// IHR_24386824306815
// MI455X (gfx1250) — matrix operand order repaired and checked by reading, not compiled, not run on hardware
//
#include <hip/hip_runtime.h>

typedef __attribute__((ext_vector_type(16))) __bf16 v16bf;
typedef __attribute__((ext_vector_type(8)))  float  v8f;

#define DEV __device__ __forceinline__

DEV int kmapA(int e, int lane) { return e + 16 * (lane >> 4); }

DEV v8f wmma_bf16(v16bf a, v16bf b, v8f c) {
  return __builtin_amdgcn_wmma_f32_16x16x32_bf16(false, a, false, b, (short)0, c, false, false);
}

DEV float lane_swap16(float x) {  // exchange lanes l <-> l^16 (SWAPX16)
  return __int_as_float(__builtin_amdgcn_ds_swizzle(__float_as_int(x), 0x401F));
}

// ---------------------------------------------------------------- pooled
__global__ void pooled_k(const float* __restrict__ ref, float* __restrict__ pooled) {
  __shared__ float red[256];
  int tid = threadIdx.x;
  const float* p = ref + (size_t)blockIdx.x * 4096;
  float s = 0.f;
  for (int i = tid; i < 4096; i += 256) s += p[i];
  red[tid] = s;
  __syncthreads();
  for (int off = 128; off > 0; off >>= 1) {
    if (tid < off) red[tid] += red[tid + off];
    __syncthreads();
  }
  if (tid == 0) pooled[blockIdx.x] = red[0] * (1.f / 4096.f);
}

// ---------------------------------------------------------------- small folds
// tmp1 = W1[:, :128] @ W1 ; tmp2 = W1[:, 128:] @ W1 ; inner = (B1+B2)@b1 + b1
__global__ void prep_k(const float* __restrict__ w1, const float* __restrict__ b1,
                       float* __restrict__ tmp1, float* __restrict__ tmp2,
                       float* __restrict__ inner) {
  int t = blockIdx.x * 256 + threadIdx.x;  // 32768
  int i = t >> 8, c = t & 255;
  float s1 = 0.f, s2 = 0.f;
  for (int j = 0; j < 128; ++j) {
    float wj = w1[j * 256 + c];
    s1 += w1[i * 256 + j] * wj;
    s2 += w1[i * 256 + 128 + j] * wj;
  }
  tmp1[i * 256 + c] = s1;
  tmp2[i * 256 + c] = s2;
  if (c == 0) {
    float s = b1[i];
    for (int j = 0; j < 128; ++j) s += (w1[i * 256 + j] + w1[i * 256 + 128 + j]) * b1[j];
    inner[i] = s;
  }
}

// M1=Am@tmp1, M2=Am@tmp2, M3=Am2@W1, M4=Am3 ; btot
__global__ void fold_k(const float* __restrict__ mw, const float* __restrict__ mb,
                       const float* __restrict__ w1, const float* __restrict__ b1,
                       const float* __restrict__ tmp1, const float* __restrict__ tmp2,
                       const float* __restrict__ inner, float* __restrict__ M,
                       float* __restrict__ btot) {
  int t = blockIdx.x * 256 + threadIdx.x;  // 65536
  int o = t >> 8, c = t & 255;
  float m0 = 0.f, m1 = 0.f, m2 = 0.f;
  for (int i = 0; i < 128; ++i) {
    float a = mw[o * 512 + i];
    m0 += a * tmp1[i * 256 + c];
    m1 += a * tmp2[i * 256 + c];
    m2 += mw[o * 512 + 128 + i] * w1[i * 256 + c];
  }
  M[(0 * 256 + o) * 256 + c] = m0;
  M[(1 * 256 + o) * 256 + c] = m1;
  M[(2 * 256 + o) * 256 + c] = m2;
  M[(3 * 256 + o) * 256 + c] = mw[o * 512 + 256 + c];
  if (c == 0) {
    float s = mb[o];
    for (int i = 0; i < 128; ++i)
      s += mw[o * 512 + i] * inner[i] + mw[o * 512 + 128 + i] * b1[i];
    btot[o] = s;
  }
}

// ---------------------------------------------------------------- fragment packers
__global__ void packM_k(const float* __restrict__ M, v16bf* __restrict__ Mfrag) {
  int t = blockIdx.x * 256 + threadIdx.x;  // 16384
  int lane = t & 31, ot = (t >> 5) & 15, cc = (t >> 9) & 7, f = t >> 12;
  int o = ot * 16 + (lane & 15);
  v16bf v;
#pragma unroll
  for (int e = 0; e < 16; ++e) {
    int c = cc * 32 + kmapA(e, lane);
    v[e] = (__bf16)M[(f * 256 + o) * 256 + c];
  }
  Mfrag[t] = v;
}

__global__ void packW_k(const float* __restrict__ ktw, const float* __restrict__ kqw,
                        v16bf* __restrict__ Wfrag) {
  int t = blockIdx.x * 256 + threadIdx.x;  // 9216
  if (t >= 9216) return;
  int lane = t & 31;
  int r = t >> 5;
  int kt = r & 1; r >>= 1;
  int cc = r & 7; r >>= 3;
  int tap = r % 9, which = r / 9;
  const float* w = which ? kqw : ktw;
  int k = kt * 16 + (lane & 15);
  int dy = tap / 3, dx = tap % 3;
  v16bf v;
#pragma unroll
  for (int e = 0; e < 16; ++e) {
    int c = cc * 32 + kmapA(e, lane);
    v[e] = (__bf16)w[((k * 256 + c) * 3 + dy) * 3 + dx];
  }
  Wfrag[t] = v;
}

__global__ void packRef_k(const float* __restrict__ ref, v16bf* __restrict__ refFrag) {
  int t = blockIdx.x * 256 + threadIdx.x;  // 262144
  int lane = t & 31, ct = (t >> 5) & 15, mc = (t >> 9) & 127, b = t >> 16;
  int c = ct * 16 + (lane & 15);
  v16bf v;
#pragma unroll
  for (int e = 0; e < 16; ++e) {
    int m = mc * 32 + kmapA(e, lane);
    v[e] = (__bf16)ref[(size_t)(b * 256 + c) * 4096 + m];
  }
  refFrag[t] = v;
}

// ---------------------------------------------------------------- conv3x3 -> key fragments
// WG: 4 output rows x 64 cols of one batch. GEMM over K = 256ch x 9 taps.
__global__ void conv_k(const float* __restrict__ x, const float* __restrict__ ref,
                       const v16bf* __restrict__ Wfrag, const float* __restrict__ ktb,
                       const float* __restrict__ kqb, v16bf* __restrict__ ktFrag,
                       v16bf* __restrict__ kqFrag) {
  __shared__ __align__(64) __bf16 xs[6 * 66 * 32];
  int h0 = blockIdx.x * 4, b = blockIdx.y, which = blockIdx.z;
  const float* src = which ? ref : x;
  const float* bias = which ? kqb : ktb;
  int tid = threadIdx.x, lane = tid & 31, wv = tid >> 5;
  const v16bf* wf = Wfrag + which * 9 * 8 * 2 * 32;

  v8f acc[2][2];
  v8f zero = {0.f, 0.f, 0.f, 0.f, 0.f, 0.f, 0.f, 0.f};
#pragma unroll
  for (int a = 0; a < 2; ++a)
#pragma unroll
    for (int k = 0; k < 2; ++k) acc[a][k] = zero;

  for (int cc = 0; cc < 8; ++cc) {
    __syncthreads();
    for (int i = tid; i < 6 * 66 * 32; i += 256) {
      int c = i & 31, pix = i >> 5;
      int ww = pix % 66, hh = pix / 66;
      int hin = h0 - 1 + hh, win = ww - 1;
      float v = 0.f;
      if ((unsigned)hin < 64u && (unsigned)win < 64u)
        v = src[(size_t)(b * 256 + cc * 32 + c) * 4096 + hin * 64 + win];
      xs[i] = (__bf16)v;
    }
    __syncthreads();
    for (int tap = 0; tap < 9; ++tap) {
      int dy = tap / 3, dx = tap % 3;
      v16bf a0 = wf[((tap * 8 + cc) * 2 + 0) * 32 + lane];
      v16bf a1 = wf[((tap * 8 + cc) * 2 + 1) * 32 + lane];
#pragma unroll
      for (int tl = 0; tl < 2; ++tl) {
        int tIdx = wv * 2 + tl;
        int hh = (tIdx >> 2) + dy;
        int ww = (tIdx & 3) * 16 + (lane & 15) + dx;
        const __bf16* pp = xs + ((hh * 66 + ww) << 5) + ((lane >> 4) << 4);
        v16bf bx = *(const v16bf*)pp;
        acc[tl][0] = wmma_bf16(a0, bx, acc[tl][0]);
        acc[tl][1] = wmma_bf16(a1, bx, acc[tl][1]);
      }
    }
  }

  // Epilogue: bias + direct fragment packing. key_t -> A layout, key_q -> B layout.
#pragma unroll
  for (int tl = 0; tl < 2; ++tl) {
    int tIdx = wv * 2 + tl;
    int ntile = (h0 + (tIdx >> 2)) * 4 + (tIdx & 3);
    size_t fi = (size_t)(b * 256 + ntile) * 32 + lane;
    if (which == 0) {
      v16bf v;
#pragma unroll
      for (int e = 0; e < 16; ++e) {
        int r = e & 7;
        float val = (e < 8) ? acc[tl][0][r] : acc[tl][1][r];
        int k = r + 8 * (lane >> 4) + 16 * (e >> 3);
        v[e] = (__bf16)(val + bias[k]);
      }
      ktFrag[fi] = v;
    } else {
      float sw0[8], sw1[8];
#pragma unroll
      for (int r = 0; r < 8; ++r) {
        sw0[r] = lane_swap16(acc[tl][0][r]);
        sw1[r] = lane_swap16(acc[tl][1][r]);
      }
      int hi = lane >> 4;
      v16bf v;
#pragma unroll
      for (int e = 0; e < 16; ++e) {
        float val = (e < 8) ? acc[tl][0][e] : acc[tl][1][e - 8];
        int k = (e & 7) + 8 * hi + 16 * (e >> 3);
        v[e] = (__bf16)(val + bias[k]);
      }
      kqFrag[fi] = v;
    }
  }
}

// ---------------------------------------------------------------- pass 1: column softmax stats
__global__ void stats_k(const v16bf* __restrict__ ktFrag, const v16bf* __restrict__ kqFrag,
                        float* __restrict__ colmax, float* __restrict__ colsum) {
  int b = blockIdx.y, lane = threadIdx.x & 31, wv = threadIdx.x >> 5;
  int mt = blockIdx.x * 8 + wv;
  v16bf bq = kqFrag[(size_t)(b * 256 + mt) * 32 + lane];
  v8f zero = {0.f, 0.f, 0.f, 0.f, 0.f, 0.f, 0.f, 0.f};
  float M = -3.0e38f, S = 0.f;
  for (int nt = 0; nt < 256; ++nt) {
    v16bf a = ktFrag[(size_t)(b * 256 + nt) * 32 + lane];
    v8f p = wmma_bf16(a, bq, zero);
    float tm = p[0];
#pragma unroll
    for (int r = 1; r < 8; ++r) tm = fmaxf(tm, p[r]);
    float nM = fmaxf(M, tm);
    float s = S * __expf(M - nM);
#pragma unroll
    for (int r = 0; r < 8; ++r) s += __expf(p[r] - nM);
    M = nM;
    S = s;
  }
  float pM = lane_swap16(M), pS = lane_swap16(S);
  float nM = fmaxf(M, pM);
  float tot = S * __expf(M - nM) + pS * __expf(pM - nM);
  if (lane < 16) {
    colmax[b * 4096 + mt * 16 + lane] = nM;
    colsum[b * 4096 + mt * 16 + lane] = tot;
  }
}

// ---------------------------------------------------------------- pass 2: val = ref @ softmax(p)
__global__ void attend_k(const v16bf* __restrict__ ktFrag, const v16bf* __restrict__ kqFrag,
                         const v16bf* __restrict__ refFrag, const float* __restrict__ colmax,
                         const float* __restrict__ colsum, float* __restrict__ val) {
  __shared__ __align__(64) __bf16 wl[8 * 16 * 32];
  int b = blockIdx.y, lane = threadIdx.x & 31, wv = threadIdx.x >> 5;
  int ntile = blockIdx.x * 8 + wv;
  __bf16* myw = wl + wv * 512;

  v16bf at = ktFrag[(size_t)(b * 256 + ntile) * 32 + lane];
  v8f zero = {0.f, 0.f, 0.f, 0.f, 0.f, 0.f, 0.f, 0.f};
  v8f acc[16];
#pragma unroll
  for (int ct = 0; ct < 16; ++ct) acc[ct] = zero;

  const v16bf* rf = refFrag + (size_t)b * 128 * 16 * 32;
  int rowbase = (lane >> 4) << 3;

  for (int mc = 0; mc < 128; ++mc) {
#pragma unroll
    for (int h = 0; h < 2; ++h) {
      int mt = mc * 2 + h;
      v16bf bq = kqFrag[(size_t)(b * 256 + mt) * 32 + lane];
      v8f p = wmma_bf16(at, bq, zero);
      int m = mt * 16 + (lane & 15);
      float cm = colmax[b * 4096 + m];
      float inv = 1.0f / colsum[b * 4096 + m];
#pragma unroll
      for (int r = 0; r < 8; ++r)
        myw[(rowbase + r) * 32 + h * 16 + (lane & 15)] = (__bf16)(__expf(p[r] - cm) * inv);
    }
    asm volatile("s_wait_dscnt 0" ::: "memory");
    v16bf bw = *(const v16bf*)(myw + ((lane & 15) << 5) + ((lane >> 4) << 4));
    const v16bf* rfm = rf + (size_t)mc * 16 * 32;
#pragma unroll
    for (int ct = 0; ct < 16; ++ct) acc[ct] = wmma_bf16(rfm[ct * 32 + lane], bw, acc[ct]);
    asm volatile("" ::: "memory");
  }

  int n = ntile * 16 + (lane & 15);
#pragma unroll
  for (int ct = 0; ct < 16; ++ct)
#pragma unroll
    for (int r = 0; r < 8; ++r) {
      int c = ct * 16 + r + rowbase;
      val[(size_t)(b * 256 + c) * 4096 + n] = acc[ct][r];
    }
}

// ---------------------------------------------------------------- fused epilogue GEMM (K=1024)
__global__ void fuse_k(const float* __restrict__ x, const float* __restrict__ val,
                       const float* __restrict__ pooled, const v16bf* __restrict__ Mfrag,
                       const float* __restrict__ btot, float* __restrict__ out) {
  __shared__ __align__(64) __bf16 feat[4 * 64 * 32];
  int b = blockIdx.y, nb = blockIdx.x;
  int tid = threadIdx.x, lane = tid & 31, wv = tid >> 5;

  v8f zero = {0.f, 0.f, 0.f, 0.f, 0.f, 0.f, 0.f, 0.f};
  v8f acc[2][4];
#pragma unroll
  for (int ot = 0; ot < 2; ++ot)
#pragma unroll
    for (int nt = 0; nt < 4; ++nt) acc[ot][nt] = zero;

  for (int cc = 0; cc < 8; ++cc) {
    __syncthreads();
    for (int i = tid; i < 2048; i += 256) {
      int cl = i & 31, nl = i >> 5;
      int c = cc * 32 + cl;
      size_t base = (size_t)(b * 256 + c) * 4096 + nb * 64 + nl;
      float xv = x[base];
      float vv = val[base];
      float pl = pooled[b * 256 + c];
      feat[(0 * 64 + nl) * 32 + cl] = (__bf16)(xv * pl);
      feat[(1 * 64 + nl) * 32 + cl] = (__bf16)fabsf(pl - xv);
      feat[(2 * 64 + nl) * 32 + cl] = (__bf16)fabsf(vv - xv);
      feat[(3 * 64 + nl) * 32 + cl] = (__bf16)xv;
    }
    __syncthreads();
#pragma unroll
    for (int f = 0; f < 4; ++f) {
#pragma unroll
      for (int nt = 0; nt < 4; ++nt) {
        const __bf16* pp = feat + ((f * 64 + nt * 16 + (lane & 15)) << 5) + ((lane >> 4) << 4);
        v16bf bx = *(const v16bf*)pp;
#pragma unroll
        for (int ot = 0; ot < 2; ++ot) {
          v16bf a = Mfrag[((f * 8 + cc) * 16 + (wv * 2 + ot)) * 32 + lane];
          acc[ot][nt] = wmma_bf16(a, bx, acc[ot][nt]);
        }
      }
    }
  }

#pragma unroll
  for (int ot = 0; ot < 2; ++ot)
#pragma unroll
    for (int nt = 0; nt < 4; ++nt)
#pragma unroll
      for (int r = 0; r < 8; ++r) {
        int o = wv * 32 + ot * 16 + r + ((lane >> 4) << 3);
        int n = nb * 64 + nt * 16 + (lane & 15);
        out[(size_t)(b * 256 + o) * 4096 + n] = acc[ot][nt][r] + btot[o];
      }
}

// ---------------------------------------------------------------- launch
extern "C" void kernel_launch(void* const* d_in, const int* in_sizes, int n_in,
                              void* d_out, int out_size, void* d_ws, size_t ws_size,
                              hipStream_t stream) {
  (void)in_sizes; (void)n_in; (void)out_size; (void)ws_size;
  const float* x   = (const float*)d_in[0];
  const float* ref = (const float*)d_in[1];
  const float* ktw = (const float*)d_in[2];
  const float* ktb = (const float*)d_in[3];
  const float* kqw = (const float*)d_in[4];
  const float* kqb = (const float*)d_in[5];
  const float* w1  = (const float*)d_in[6];
  const float* b1  = (const float*)d_in[7];
  const float* mw  = (const float*)d_in[8];
  const float* mb  = (const float*)d_in[9];
  float* out = (float*)d_out;

  char* ws = (char*)d_ws;
  size_t cur = 0;
  auto alloc = [&](size_t bytes) {
    size_t p = cur;
    cur += (bytes + 255) & ~(size_t)255;
    return (void*)(ws + p);
  };
  float* pooled  = (float*)alloc(1024 * 4);
  float* tmp1    = (float*)alloc(128 * 256 * 4);
  float* tmp2    = (float*)alloc(128 * 256 * 4);
  float* inner   = (float*)alloc(128 * 4);
  float* Mbuf    = (float*)alloc(4 * 256 * 256 * 4);
  float* btot    = (float*)alloc(256 * 4);
  v16bf* Mfrag   = (v16bf*)alloc(16384 * 32);
  v16bf* Wfrag   = (v16bf*)alloc(9216 * 32);
  v16bf* ktFrag  = (v16bf*)alloc(32768 * 32);
  v16bf* kqFrag  = (v16bf*)alloc(32768 * 32);
  v16bf* refFrag = (v16bf*)alloc(262144 * 32);
  float* colmax  = (float*)alloc(16384 * 4);
  float* colsum  = (float*)alloc(16384 * 4);
  float* val     = (float*)alloc((size_t)4 * 256 * 4096 * 4);

  pooled_k<<<1024, 256, 0, stream>>>(ref, pooled);
  prep_k<<<128, 256, 0, stream>>>(w1, b1, tmp1, tmp2, inner);
  fold_k<<<256, 256, 0, stream>>>(mw, mb, w1, b1, tmp1, tmp2, inner, Mbuf, btot);
  packM_k<<<64, 256, 0, stream>>>(Mbuf, Mfrag);
  packW_k<<<36, 256, 0, stream>>>(ktw, kqw, Wfrag);
  packRef_k<<<1024, 256, 0, stream>>>(ref, refFrag);
  conv_k<<<dim3(16, 4, 2), 256, 0, stream>>>(x, ref, Wfrag, ktb, kqb, ktFrag, kqFrag);
  stats_k<<<dim3(32, 4), 256, 0, stream>>>(ktFrag, kqFrag, colmax, colsum);
  attend_k<<<dim3(32, 4), 256, 0, stream>>>(ktFrag, kqFrag, refFrag, colmax, colsum, val);
  fuse_k<<<dim3(64, 4), 256, 0, stream>>>(x, val, pooled, Mfrag, btot, out);
}
